// SpiralMeshReader_89653147336992
// MI455X (gfx1250) — compile-verified, matrix operand order checked by reading, not run on hardware
//
#include <hip/hip_runtime.h>
#include <hip/hip_bf16.h>

#define NN   65536
#define EE   1048576
#define HHD  64
#define OO   32
#define BB   32
#define NPG  2048
#define EPSF 1e-5f
#define SLOPE 0.01f

typedef __attribute__((ext_vector_type(16))) __bf16 v16bf;
typedef __attribute__((ext_vector_type(8)))  float  v8f;

// ---------------- utility ----------------
__global__ void k_zero(float* __restrict__ p, int n) {
  int i = blockIdx.x * blockDim.x + threadIdx.x;
  if (i < n) p[i] = 0.f;
}

// ---------------- degrees ----------------
__global__ void k_degrees(const int* __restrict__ src, const int* __restrict__ dst,
                          float* __restrict__ degO, float* __restrict__ degI) {
  int e = blockIdx.x * blockDim.x + threadIdx.x;
  if (e < EE) {
    atomicAdd(&degO[src[e]], 1.f);
    atomicAdd(&degI[dst[e]], 1.f);
  }
}

__global__ void k_invdeg(const float* __restrict__ degO, const float* __restrict__ degI,
                         float* __restrict__ invO, float* __restrict__ invI) {
  int i = blockIdx.x * blockDim.x + threadIdx.x;
  if (i < NN) {
    invO[i] = rsqrtf(fmaxf(degO[i], 1.f));
    invI[i] = rsqrtf(fmaxf(degI[i], 1.f));
  }
}

// ---------------- bf16 WMMA GEMM: Y[N,64] = (X * scale[:,None]) @ W[64,64] ----------------
// Block = 256 thr = 8 waves; block tile 32 rows x 64 cols; wave tile 16x16; K=64 in two x32 steps.
__global__ void __launch_bounds__(256)
k_gemm_bf16(const float* __restrict__ X, const float* __restrict__ scale,
            const float* __restrict__ W, float* __restrict__ Y) {
  const int lane = threadIdx.x & 31;
  const int wave = threadIdx.x >> 5;
  const int hh   = lane >> 4;      // lane half selector
  const int r    = lane & 15;
  const int rowBase = blockIdx.x * 32 + (wave & 1) * 16;
  const int colBase = (wave >> 1) * 16;

  v8f c = {};
#pragma unroll
  for (int kc = 0; kc < 64; kc += 32) {
    v16bf a, b;
    const int   arow = rowBase + r;
    const float s    = scale[arow];
    const float* xr  = X + (size_t)arow * HHD;
#pragma unroll
    for (int i = 0; i < 16; ++i) {
      // 16-bit A 16x32 layout: lane<16 -> K {0..7,16..23}; lane>=16 -> K {8..15,24..31}
      const int k = kc + ((i < 8) ? 0 : 16) + hh * 8 + (i & 7);
      a[i] = (__bf16)(xr[k] * s);
      b[i] = (__bf16)(W[k * HHD + colBase + r]);
    }
    c = __builtin_amdgcn_wmma_f32_16x16x32_bf16(false, a, false, b, (short)0, c,
                                                false, false);
  }
#pragma unroll
  for (int v = 0; v < 8; ++v) {
    const int row = rowBase + v + 8 * hh;           // D: M = v + 8*half, N = lane&15
    Y[(size_t)row * HHD + colBase + r] = c[v];
  }
}

// ---------------- edge scatter: agg[dst] += ew * hW[src] ----------------
// one wave per edge; each lane moves a float2 (64 ch / 32 lanes)
__global__ void __launch_bounds__(256)
k_scatter(const float* __restrict__ hW, const int* __restrict__ src,
          const int* __restrict__ dst, const float* __restrict__ ew,
          float* __restrict__ agg) {
  const int lane = threadIdx.x & 31;
  const int gw   = (blockIdx.x * blockDim.x + threadIdx.x) >> 5;
  const int nw   = (gridDim.x * blockDim.x) >> 5;
  for (int e = gw; e < EE; e += nw) {
    const int   s = src[e];
    const int   d = dst[e];
    const float w = ew[e];
    if (e + nw < EE)
      __builtin_prefetch(&hW[(size_t)src[e + nw] * HHD], 0, 0); // global_prefetch_b8
    const float2 m = ((const float2*)(hW + (size_t)s * HHD))[lane];
    float* ad = agg + (size_t)d * HHD + lane * 2;
    atomicAdd(ad,     m.x * w);
    atomicAdd(ad + 1, m.y * w);
  }
}

// ---------------- fused leaky + column sum/sumsq reduction ----------------
__global__ void __launch_bounds__(256)
k_stats(const float* __restrict__ agg, const float* __restrict__ invI,
        float* __restrict__ colsum, float* __restrict__ colsq) {
  __shared__ float ls[256], lq[256];
  const int j  = threadIdx.x & 63;
  const int rr = threadIdx.x >> 6;
  float s = 0.f, q = 0.f;
  const int base = blockIdx.x * 256;
  for (int n = base + rr; n < base + 256; n += 4) {
    float y = agg[(size_t)n * HHD + j] * invI[n];
    y = (y >= 0.f) ? y : SLOPE * y;
    s += y; q += y * y;
  }
  ls[threadIdx.x] = s; lq[threadIdx.x] = q;
  __syncthreads();
  if (rr == 0) {
    s = ls[j] + ls[j + 64] + ls[j + 128] + ls[j + 192];
    q = lq[j] + lq[j + 64] + lq[j + 128] + lq[j + 192];
    atomicAdd(&colsum[j], s);
    atomicAdd(&colsq[j], q);
  }
}

__global__ void k_finstats(const float* __restrict__ colsum, const float* __restrict__ colsq,
                           const float* __restrict__ ga,
                           float* __restrict__ shift, float* __restrict__ rstd) {
  int j = threadIdx.x;
  if (j < HHD) {
    const float mean = colsum[j] * (1.f / NN);
    const float ex2  = colsq[j]  * (1.f / NN);
    const float a    = ga[j];
    const float var  = ex2 - 2.f * a * mean * mean + a * a * mean * mean;
    shift[j] = a * mean;
    rstd[j]  = rsqrtf(var + EPSF);
  }
}

__global__ void __launch_bounds__(256)
k_norm(const float* __restrict__ agg, const float* __restrict__ invI,
       const float* __restrict__ shift, const float* __restrict__ rstd,
       const float* __restrict__ gw, const float* __restrict__ gb,
       float* __restrict__ out) {
  int t = blockIdx.x * blockDim.x + threadIdx.x;
  if (t < NN * HHD) {
    const int n = t >> 6, j = t & 63;
    float y = agg[t] * invI[n];
    y = (y >= 0.f) ? y : SLOPE * y;
    out[t] = gw[j] * ((y - shift[j]) * rstd[j]) + gb[j];
  }
}

// ---------------- per-graph mean pooling (contiguous 2048-node graphs) ----------------
__global__ void __launch_bounds__(256)
k_pool(const float* __restrict__ h, float* __restrict__ pooled) {
  __shared__ float ls[256];
  const int g  = blockIdx.x;
  const int j  = threadIdx.x & 63;
  const int rr = threadIdx.x >> 6;
  float s = 0.f;
  for (int n = g * NPG + rr; n < (g + 1) * NPG; n += 4)
    s += h[(size_t)n * HHD + j];
  ls[threadIdx.x] = s;
  __syncthreads();
  if (rr == 0)
    pooled[g * HHD + j] =
        (ls[j] + ls[j + 64] + ls[j + 128] + ls[j + 192]) * (1.f / NPG);
}

// ---------------- classifier: out[B,O] = pooled[B,H] @ Wc[O,H]^T ----------------
__global__ void k_classify(const float* __restrict__ pooled, const float* __restrict__ Wc,
                           float* __restrict__ out) {
  const int t = threadIdx.x;              // 1024 threads = B*O
  const int b = t >> 5, o = t & 31;
  const float* p = pooled + b * HHD;
  const float* w = Wc + o * HHD;
  float s = 0.f;
#pragma unroll
  for (int k = 0; k < HHD; ++k) s += p[k] * w[k];
  out[b * OO + o] = s;
}

// ---------------- driver ----------------
extern "C" void kernel_launch(void* const* d_in, const int* in_sizes, int n_in,
                              void* d_out, int out_size, void* d_ws, size_t ws_size,
                              hipStream_t stream) {
  const float* x    = (const float*)d_in[0];
  const float* ew   = (const float*)d_in[1];
  const float* W1   = (const float*)d_in[2];
  const float* W2   = (const float*)d_in[3];
  const float* Wc   = (const float*)d_in[4];
  const float* gn1w = (const float*)d_in[5];
  const float* gn1b = (const float*)d_in[6];
  const float* gn1a = (const float*)d_in[7];
  const float* gn2w = (const float*)d_in[8];
  const float* gn2b = (const float*)d_in[9];
  const float* gn2a = (const float*)d_in[10];
  const int*   src  = (const int*)d_in[11];
  const int*   dst  = (const int*)d_in[12];
  float*       out  = (float*)d_out;

  float* ws = (float*)d_ws;
  float* degO   = ws;                 ws += NN;
  float* degI   = ws;                 ws += NN;
  float* invO   = ws;                 ws += NN;
  float* invI   = ws;                 ws += NN;
  float* bufA   = ws;                 ws += (size_t)NN * HHD;
  float* bufB   = ws;                 ws += (size_t)NN * HHD;
  float* colsum = ws;                 ws += HHD;
  float* colsq  = ws;                 ws += HHD;
  float* shiftv = ws;                 ws += HHD;
  float* rstdv  = ws;                 ws += HHD;
  float* pooled = ws;                 ws += BB * HHD;

  const int T = 256;
  const int NH = NN * HHD;

  // degrees + norms
  k_zero<<<(2 * NN + T - 1) / T, T, 0, stream>>>(degO, 2 * NN); // degO+degI contiguous
  k_degrees<<<EE / T, T, 0, stream>>>(src, dst, degO, degI);
  k_invdeg<<<NN / T, T, 0, stream>>>(degO, degI, invO, invI);

  // ----- layer 1 -----
  k_gemm_bf16<<<NN / 32, T, 0, stream>>>(x, invO, W1, bufA);
  k_zero<<<NH / T, T, 0, stream>>>(bufB, NH);
  k_scatter<<<2048, T, 0, stream>>>(bufA, src, dst, ew, bufB);
  k_zero<<<1, T, 0, stream>>>(colsum, 2 * HHD);                 // colsum+colsq contiguous
  k_stats<<<NN / 256, T, 0, stream>>>(bufB, invI, colsum, colsq);
  k_finstats<<<1, 64, 0, stream>>>(colsum, colsq, gn1a, shiftv, rstdv);
  k_norm<<<NH / T, T, 0, stream>>>(bufB, invI, shiftv, rstdv, gn1w, gn1b, bufA);

  // ----- layer 2 -----
  k_gemm_bf16<<<NN / 32, T, 0, stream>>>(bufA, invO, W2, bufB);
  k_zero<<<NH / T, T, 0, stream>>>(bufA, NH);
  k_scatter<<<2048, T, 0, stream>>>(bufB, src, dst, ew, bufA);
  k_zero<<<1, T, 0, stream>>>(colsum, 2 * HHD);
  k_stats<<<NN / 256, T, 0, stream>>>(bufA, invI, colsum, colsq);
  k_finstats<<<1, 64, 0, stream>>>(colsum, colsq, gn2a, shiftv, rstdv);
  k_norm<<<NH / T, T, 0, stream>>>(bufA, invI, shiftv, rstdv, gn2w, gn2b, bufB);

  // ----- pool + classify -----
  k_pool<<<BB, T, 0, stream>>>(bufB, pooled);
  k_classify<<<1, BB * OO, 0, stream>>>(pooled, Wc, out);
}
